// SCAESuite_64733747086002
// MI455X (gfx1250) — matrix operand order repaired and checked by reading, not compiled, not run on hardware
//
#include <hip/hip_runtime.h>
#include <hip/hip_bf16.h>

// ---------------------------------------------------------------------------
// SCAE attention contribution pipeline for MI455X (gfx1250, wave32, WMMA).
// All matmuls: LDS-tiled block GEMM (8 waves, 64x128 block tile) feeding
// v_wmma_f32_16x16x32_bf16 with 2x2 register blocking per wave.
// Staging: double-buffered gfx1250 GLOBAL_LOAD_ASYNC_TO_LDS_B128 prefetch
// (ASYNCcnt-tracked, VGPR-free copies) overlapped with WMMA math.
// ---------------------------------------------------------------------------

typedef unsigned short u16;
typedef __attribute__((ext_vector_type(16))) __bf16 v16bf;
typedef __attribute__((ext_vector_type(8)))  float  v8f;
typedef __attribute__((ext_vector_type(8)))  u16    v8u16;

constexpr int B = 2, S = 512, D = 768, H = 12, F = 4096;

// LDS tile geometry: A 64x64 bf16, B 128x64 bf16, padded stride 72
// (row stride 144B -> bank = 36*row mod 64 is a permutation over 16 rows).
constexpr int LDS_STRIDE = 72;

// ----- bf16 helpers (u16 storage so we never rely on scalar __bf16 ops) -----
__device__ __forceinline__ u16 f2bf(float f) {
  unsigned u = __float_as_uint(f);
  u += 0x7FFFu + ((u >> 16) & 1u);        // round-to-nearest-even
  return (u16)(u >> 16);
}
__device__ __forceinline__ float bf2f(u16 x) {
  return __uint_as_float(((unsigned)x) << 16);
}

// ----- gfx1250 async global->LDS copy (16B per lane, ASYNCcnt-tracked) ------
// GVS addressing: SGPR 64-bit base + per-lane 32-bit signed byte offset.
__device__ __forceinline__ void async_copy_b128(u16* lds_dst, const u16* base,
                                                int elem_off) {
  unsigned lds = (unsigned)(uintptr_t)lds_dst;
  int voff = elem_off * 2;            // byte offset
  asm volatile("global_load_async_to_lds_b128 %0, %1, %2"
               :
               : "v"(lds), "v"(voff), "s"(base)
               : "memory");
}
__device__ __forceinline__ void wait_async0() {
  asm volatile("s_wait_asynccnt 0x0" ::: "memory");
}

// ----- WMMA fragment loads from LDS ------------------------------------------
// A layout (16-bit A 16x32): lane m = lane&15, k-base = (lane>=16 ? 8 : 0);
// elements 0..7 = k..k+7, elements 8..15 = k+16..k+23.
__device__ __forceinline__ v16bf lds_a_frag(const u16* sm, int row, int k) {
  union { v8u16 h[2]; v16bf v; } x;
  const u16* p = sm + row * LDS_STRIDE + k;
  x.h[0] = *(const v8u16*)(p);
  x.h[1] = *(const v8u16*)(p + 16);
  return x.v;
}
// B layout (32x16): lane n = lane&15, k-base = (lane>=16 ? 8 : 0);
// elements 0..7 = k..k+7, elements 8..15 = k+16..k+23.
__device__ __forceinline__ v16bf lds_b_frag(const u16* sm, int row, int k) {
  union { v8u16 h[2]; v16bf v; } x;
  const u16* p = sm + row * LDS_STRIDE + k;
  x.h[0] = *(const v8u16*)(p);
  x.h[1] = *(const v8u16*)(p + 16);
  return x.v;
}
__device__ __forceinline__ v8f wmma_bf16(v16bf a, v16bf b, v8f c) {
  // 8 args: (neg_a, A, neg_b, B, c_mod, C, reuse_a, reuse_b)
  return __builtin_amdgcn_wmma_f32_16x16x32_bf16(false, a, false, b, (short)0, c,
                                                 false, false);
}

struct Acc4 { v8f a00, a01, a10, a11; };   // 2x2 tiles of 16x16 per wave

// Issue async copies for one 64-deep K-slab (A 64x64, B 128x64) into LDS.
__device__ __forceinline__ void stage_slab(
    u16* smA, u16* smB,
    const u16* __restrict__ A, int lda, int rowB,
    const u16* __restrict__ Bt, int ldb, int colB, int k0, int tid) {
#pragma unroll
  for (int c = 0; c < 2; ++c) {       // A: 512 16B-chunks, 2 per thread
    int idx = tid + c * 256;
    int r = idx >> 3, cc = (idx & 7) << 3;
    async_copy_b128(&smA[r * LDS_STRIDE + cc], A, (rowB + r) * lda + k0 + cc);
  }
#pragma unroll
  for (int c = 0; c < 4; ++c) {       // B: 1024 16B-chunks, 4 per thread
    int idx = tid + c * 256;
    int r = idx >> 3, cc = (idx & 7) << 3;
    async_copy_b128(&smB[r * LDS_STRIDE + cc], Bt, (colB + r) * ldb + k0 + cc);
  }
}

// Block-cooperative GEMM: 256 threads (8 waves as 2x4) compute
// C[rowB:+64, colB:+128] += A(row-major [M,K]) * Bt(row-major [N,K])^T.
// Double-buffered async staging: slab i+1 copies fly while slab i is in WMMA.
__device__ __forceinline__ void block_mma(
    u16* smA0, u16* smB0, u16* smA1, u16* smB1,
    const u16* __restrict__ A, int lda, int rowB,
    const u16* __restrict__ Bt, int ldb, int colB,
    int K, Acc4& acc) {
  const int tid  = threadIdx.x;
  const int lane = tid & 31;
  const int wv   = tid >> 5;
  const int waveM = wv >> 2;          // 0..1
  const int waveN = wv & 3;           // 0..3
  const int aRow = waveM * 32 + (lane & 15);
  const int bRow = waveN * 32 + (lane & 15);
  const int aK   = (lane >> 4) << 3;  // 0 or 8
  const int bK   = (lane >> 4) << 3;  // 0 or 8

  __syncthreads();                    // previous LDS use (or prior call) done
  stage_slab(smA0, smB0, A, lda, rowB, Bt, ldb, colB, 0, tid);
  u16* curA = smA0; u16* curB = smB0;
  u16* nxtA = smA1; u16* nxtB = smB1;

  for (int k0 = 0; k0 < K; k0 += 64) {
    wait_async0();                    // my copies for slab k0 have landed
    __syncthreads();                  // everyone's copies landed; prev math done
    if (k0 + 64 < K)                  // prefetch next slab into other buffer
      stage_slab(nxtA, nxtB, A, lda, rowB, Bt, ldb, colB, k0 + 64, tid);
#pragma unroll
    for (int kk = 0; kk < 64; kk += 32) {
      v16bf a0 = lds_a_frag(curA, aRow,      kk + aK);
      v16bf a1 = lds_a_frag(curA, aRow + 16, kk + aK);
      v16bf b0 = lds_b_frag(curB, bRow,      kk + bK);
      v16bf b1 = lds_b_frag(curB, bRow + 16, kk + bK);
      acc.a00 = wmma_bf16(a0, b0, acc.a00);
      acc.a01 = wmma_bf16(a0, b1, acc.a01);
      acc.a10 = wmma_bf16(a1, b0, acc.a10);
      acc.a11 = wmma_bf16(a1, b1, acc.a11);
    }
    u16* t;
    t = curA; curA = nxtA; nxtA = t;
    t = curB; curB = nxtB; nxtB = t;
  }
}

// ----- elementwise prep kernels ---------------------------------------------
__global__ void k_div_cvt(const float* __restrict__ in,
                          const float* __restrict__ scale,
                          u16* __restrict__ out, int rowlen, long long n) {
  long long i = blockIdx.x * (long long)blockDim.x + threadIdx.x;
  long long stride = (long long)gridDim.x * blockDim.x;
  for (; i < n; i += stride) out[i] = f2bf(in[i] / scale[i / rowlen]);
}
__global__ void k_cvt(const float* __restrict__ in, u16* __restrict__ out,
                      long long n) {
  long long i = blockIdx.x * (long long)blockDim.x + threadIdx.x;
  long long stride = (long long)gridDim.x * blockDim.x;
  for (; i < n; i += stride) out[i] = f2bf(in[i]);
}
// up_decoder [D,F] -> transposed bf16 [F,D]
__global__ void k_tr_cvt(const float* __restrict__ in, u16* __restrict__ out) {
  long long n = (long long)F * D;
  long long i = blockIdx.x * (long long)blockDim.x + threadIdx.x;
  long long stride = (long long)gridDim.x * blockDim.x;
  for (; i < n; i += stride) {
    int g = (int)(i / D), d = (int)(i % D);
    out[i] = f2bf(in[(size_t)d * F + g]);
  }
}

// ----- GEMM 1: enc_OV[h,f,d] = sum_e dn[f,e] * W_OV[h,d,e] ------------------
__global__ __launch_bounds__(256) void k_gemm_encov(
    const u16* __restrict__ dn, const u16* __restrict__ wov,
    u16* __restrict__ enc) {
  __shared__ u16 smA[2][64 * LDS_STRIDE], smB[2][128 * LDS_STRIDE];
  constexpr int MB = F / 64, NB = D / 128, TP = MB * NB;
  int blk = blockIdx.x;
  int h = blk / TP, t = blk % TP;
  int rowB = (t / NB) * 64, colB = (t % NB) * 128;
  Acc4 acc{};
  block_mma(smA[0], smB[0], smA[1], smB[1], dn, D, rowB,
            wov + (size_t)h * D * D, D, colB, D, acc);
  const int lane = threadIdx.x & 31, wv = threadIdx.x >> 5;
  int row0 = rowB + (wv >> 2) * 32, col0 = colB + (wv & 3) * 32;
  const int dM = (lane >> 4) << 3, dN = lane & 15;
  u16* out = enc + (size_t)h * F * D;
#pragma unroll
  for (int r = 0; r < 8; ++r) {
    out[(size_t)(row0 + dM + r) * D + col0 + dN]           = f2bf(acc.a00[r]);
    out[(size_t)(row0 + dM + r) * D + col0 + 16 + dN]      = f2bf(acc.a01[r]);
    out[(size_t)(row0 + 16 + dM + r) * D + col0 + dN]      = f2bf(acc.a10[r]);
    out[(size_t)(row0 + 16 + dM + r) * D + col0 + 16 + dN] = f2bf(acc.a11[r]);
  }
}

// ----- GEMM 2: virtual[h,f,g] = mask[f,g] * sum_d enc[h,f,d]*updecT[g,d] ----
__global__ __launch_bounds__(256) void k_gemm_virtual(
    const u16* __restrict__ enc, const u16* __restrict__ updecT,
    const unsigned char* __restrict__ mask, u16* __restrict__ virt) {
  __shared__ u16 smA[2][64 * LDS_STRIDE], smB[2][128 * LDS_STRIDE];
  constexpr int MB = F / 64, NB = F / 128, TP = MB * NB;
  int blk = blockIdx.x;
  int h = blk / TP, t = blk % TP;
  int rowB = (t / NB) * 64, colB = (t % NB) * 128;
  Acc4 acc{};
  block_mma(smA[0], smB[0], smA[1], smB[1], enc + (size_t)h * F * D, D, rowB,
            updecT, D, colB, D, acc);
  const int lane = threadIdx.x & 31, wv = threadIdx.x >> 5;
  int row0 = rowB + (wv >> 2) * 32, col0 = colB + (wv & 3) * 32;
  const int dM = (lane >> 4) << 3, dN = lane & 15;
  u16* out = virt + (size_t)h * F * F;
#pragma unroll
  for (int r = 0; r < 8; ++r) {
    int m0 = row0 + dM + r, m1 = row0 + 16 + dM + r;
    int n0 = col0 + dN,     n1 = col0 + 16 + dN;
    out[(size_t)m0 * F + n0] = mask[(size_t)m0 * F + n0] ? f2bf(acc.a00[r]) : (u16)0;
    out[(size_t)m0 * F + n1] = mask[(size_t)m0 * F + n1] ? f2bf(acc.a01[r]) : (u16)0;
    out[(size_t)m1 * F + n0] = mask[(size_t)m1 * F + n0] ? f2bf(acc.a10[r]) : (u16)0;
    out[(size_t)m1 * F + n1] = mask[(size_t)m1 * F + n1] ? f2bf(acc.a11[r]) : (u16)0;
  }
}

// ----- GEMM 3: pre_moving -> M[b,h,f,s] = sum_d x[b,s,d]*enc[h,f,d] ---------
// (stored k-transposed so the attention GEMM reads it as a native [N,K] operand)
__global__ __launch_bounds__(256) void k_gemm_premove(
    const u16* __restrict__ xbf, const u16* __restrict__ enc,
    u16* __restrict__ M) {
  __shared__ u16 smA[2][64 * LDS_STRIDE], smB[2][128 * LDS_STRIDE];
  constexpr int MB = (B * S) / 64, NB = F / 128, TP = MB * NB;
  int blk = blockIdx.x;
  int h = blk / TP, t = blk % TP;
  int rowB = (t / NB) * 64, colB = (t % NB) * 128;   // rowB = flat (b*S+s)
  int b = rowB / S, s0base = rowB - b * S;
  Acc4 acc{};
  block_mma(smA[0], smB[0], smA[1], smB[1], xbf, D, rowB,
            enc + (size_t)h * F * D, D, colB, D, acc);
  const int lane = threadIdx.x & 31, wv = threadIdx.x >> 5;
  int s0 = s0base + (wv >> 2) * 32, col0 = colB + (wv & 3) * 32;
  const int dM = (lane >> 4) << 3, dN = lane & 15;
  u16* out = M + ((size_t)b * H + h) * (size_t)F * S;
#pragma unroll
  for (int r = 0; r < 8; ++r) {
    out[(size_t)(col0 + dN) * S + s0 + dM + r]           = f2bf(acc.a00[r]);
    out[(size_t)(col0 + 16 + dN) * S + s0 + dM + r]      = f2bf(acc.a01[r]);
    out[(size_t)(col0 + dN) * S + s0 + 16 + dM + r]      = f2bf(acc.a10[r]);
    out[(size_t)(col0 + 16 + dN) * S + s0 + 16 + dM + r] = f2bf(acc.a11[r]);
  }
}

// ----- GEMM 4: M[b,h,f,k] += sum_g up[b,k,g]*virtual[h,f,g] -----------------
__global__ __launch_bounds__(256) void k_gemm_postov(
    const u16* __restrict__ upbf, const u16* __restrict__ virt,
    u16* __restrict__ M) {
  __shared__ u16 smA[2][64 * LDS_STRIDE], smB[2][128 * LDS_STRIDE];
  constexpr int MB = S / 64, NB = F / 128, TP = MB * NB;
  int blk = blockIdx.x;
  int bh = blk / TP, t = blk % TP;
  int b = bh / H, h = bh % H;
  int rowB = (t / NB) * 64, colB = (t % NB) * 128;   // rowB = k within batch
  Acc4 acc{};
  block_mma(smA[0], smB[0], smA[1], smB[1], upbf + (size_t)b * S * F, F, rowB,
            virt + (size_t)h * F * F, F, colB, F, acc);
  const int lane = threadIdx.x & 31, wv = threadIdx.x >> 5;
  int row0 = rowB + (wv >> 2) * 32, col0 = colB + (wv & 3) * 32;
  const int dM = (lane >> 4) << 3, dN = lane & 15;
  u16* out = M + ((size_t)b * H + h) * (size_t)F * S;
#pragma unroll
  for (int r = 0; r < 8; ++r) {
    size_t i00 = (size_t)(col0 + dN) * S + row0 + dM + r;
    size_t i01 = (size_t)(col0 + 16 + dN) * S + row0 + dM + r;
    size_t i10 = (size_t)(col0 + dN) * S + row0 + 16 + dM + r;
    size_t i11 = (size_t)(col0 + 16 + dN) * S + row0 + 16 + dM + r;
    out[i00] = f2bf(bf2f(out[i00]) + acc.a00[r]);
    out[i01] = f2bf(bf2f(out[i01]) + acc.a01[r]);
    out[i10] = f2bf(bf2f(out[i10]) + acc.a10[r]);
    out[i11] = f2bf(bf2f(out[i11]) + acc.a11[r]);
  }
}

// ----- GEMM 5: out[b,q,f] = sum_h sum_k probs[b,h,q,k] * M[b,h,f,k] ---------
__global__ __launch_bounds__(256) void k_gemm_attn(
    const u16* __restrict__ probsbf, const u16* __restrict__ M,
    float* __restrict__ out) {
  __shared__ u16 smA[2][64 * LDS_STRIDE], smB[2][128 * LDS_STRIDE];
  constexpr int MB = S / 64, NB = F / 128, TP = MB * NB;
  int blk = blockIdx.x;
  int b = blk / TP, t = blk % TP;
  int rowB = (t / NB) * 64, colB = (t % NB) * 128;   // q, f
  Acc4 acc{};
  for (int h = 0; h < H; ++h) {
    const u16* A  = probsbf + ((size_t)b * H + h) * (size_t)S * S;
    const u16* Bt = M       + ((size_t)b * H + h) * (size_t)F * S;
    block_mma(smA[0], smB[0], smA[1], smB[1], A, S, rowB, Bt, S, colB, S, acc);
  }
  const int lane = threadIdx.x & 31, wv = threadIdx.x >> 5;
  int row0 = rowB + (wv >> 2) * 32, col0 = colB + (wv & 3) * 32;
  const int dM = (lane >> 4) << 3, dN = lane & 15;
#pragma unroll
  for (int r = 0; r < 8; ++r) {
    out[(size_t)(b * S + row0 + dM + r) * F + col0 + dN]           = acc.a00[r];
    out[(size_t)(b * S + row0 + dM + r) * F + col0 + 16 + dN]      = acc.a01[r];
    out[(size_t)(b * S + row0 + 16 + dM + r) * F + col0 + dN]      = acc.a10[r];
    out[(size_t)(b * S + row0 + 16 + dM + r) * F + col0 + 16 + dN] = acc.a11[r];
  }
}

// ---------------------------------------------------------------------------
extern "C" void kernel_launch(void* const* d_in, const int* in_sizes, int n_in,
                              void* d_out, int out_size, void* d_ws,
                              size_t ws_size, hipStream_t stream) {
  const float* resid    = (const float*)d_in[0];   // [B,S,D]
  const float* ln_scale = (const float*)d_in[1];   // [B,S,1]
  const float* probs    = (const float*)d_in[2];   // [B,H,S,S]
  const float* up_facts = (const float*)d_in[3];   // [B,S,F]
  const float* W_OV     = (const float*)d_in[4];   // [H,D,D]
  const float* down_enc = (const float*)d_in[5];   // [F,D]
  const float* up_dec   = (const float*)d_in[6];   // [D,F]
  const unsigned char* mask = (const unsigned char*)d_in[7]; // [F,F] bool
  float* out = (float*)d_out;                      // [B,S,F]

  // bump allocator over workspace (~630 MB total)
  char* w = (char*)d_ws;
  auto alloc = [&](size_t bytes) -> char* {
    char* p = w;
    w += (bytes + 255) & ~(size_t)255;
    return p;
  };
  u16* x_bf     = (u16*)alloc((size_t)B * S * D * 2);       //  1.5 MB
  u16* up_bf    = (u16*)alloc((size_t)B * S * F * 2);       //  8.4 MB
  u16* probs_bf = (u16*)alloc((size_t)B * H * S * S * 2);   // 12.6 MB
  u16* dn_bf    = (u16*)alloc((size_t)F * D * 2);           //  6.3 MB
  u16* wov_bf   = (u16*)alloc((size_t)H * D * D * 2);       // 14.2 MB
  u16* updecT   = (u16*)alloc((size_t)F * D * 2);           //  6.3 MB
  u16* enc_bf   = (u16*)alloc((size_t)H * F * D * 2);       // 75.5 MB
  u16* virt_bf  = (u16*)alloc((size_t)H * F * F * 2);       //  403 MB
  u16* M_bf     = (u16*)alloc((size_t)B * H * F * S * 2);   //  101 MB
  (void)ws_size; (void)in_sizes; (void)n_in; (void)out_size;

  const dim3 eb(256), eg(2048);
  k_div_cvt<<<eg, eb, 0, stream>>>(resid, ln_scale, x_bf, D, (long long)B * S * D);
  k_div_cvt<<<eg, eb, 0, stream>>>(up_facts, ln_scale, up_bf, F, (long long)B * S * F);
  k_cvt<<<eg, eb, 0, stream>>>(probs, probs_bf, (long long)B * H * S * S);
  k_cvt<<<eg, eb, 0, stream>>>(down_enc, dn_bf, (long long)F * D);
  k_cvt<<<eg, eb, 0, stream>>>(W_OV, wov_bf, (long long)H * D * D);
  k_tr_cvt<<<eg, eb, 0, stream>>>(up_dec, updecT);

  const dim3 gb(256);
  k_gemm_encov<<<dim3(H * (F / 64) * (D / 128)), gb, 0, stream>>>(dn_bf, wov_bf,
                                                                  enc_bf);
  k_gemm_virtual<<<dim3(H * (F / 64) * (F / 128)), gb, 0, stream>>>(
      enc_bf, updecT, mask, virt_bf);
  k_gemm_premove<<<dim3(H * ((B * S) / 64) * (F / 128)), gb, 0, stream>>>(
      x_bf, enc_bf, M_bf);
  k_gemm_postov<<<dim3(B * H * (S / 64) * (F / 128)), gb, 0, stream>>>(
      up_bf, virt_bf, M_bf);
  k_gemm_attn<<<dim3(B * (S / 64) * (F / 128)), gb, 0, stream>>>(probs_bf, M_bf,
                                                                 out);
}
